// AnchoredBatch_11544872091858
// MI455X (gfx1250) — matrix operand order repaired and checked by reading, not compiled, not run on hardware
//
#include <hip/hip_runtime.h>
#include <hip/hip_bf16.h>

// ---------------------------------------------------------------------------
// Problem constants (from reference)
// ---------------------------------------------------------------------------
#define NEXP   4
#define INF    128          // IN features (K)
#define OUTF   128          // OUT features (N)
#define BROWS  131072       // rows per expert
#define WSZ    (NEXP * INF * OUTF)   // 65536 elements per split-weight plane

typedef __attribute__((ext_vector_type(16))) __bf16          v16bf;
typedef __attribute__((ext_vector_type(8)))  float           v8f;
typedef __attribute__((ext_vector_type(4)))  float           fx4;   // clang ext vector (not HIP_vector_type)
typedef __attribute__((ext_vector_type(16))) unsigned short  v16u;

// round-to-nearest-even float -> bf16 bits (inputs are finite; no NaN path)
static __device__ __forceinline__ unsigned short bf16_rne(float f) {
    unsigned u = __float_as_uint(f);
    u += 0x7FFFu + ((u >> 16) & 1u);
    return (unsigned short)(u >> 16);
}
static __device__ __forceinline__ float bf16_tof(unsigned short h) {
    return __uint_as_float(((unsigned)h) << 16);
}

// ---------------------------------------------------------------------------
// Kernel 1: fold r (row-scale) and s (col-scale) into a per-expert weight,
// split it into bf16 hi/lo planes, store TRANSPOSED ([e][o][i], i contiguous)
// so WMMA B-fragments are contiguous 32B per lane.
// ---------------------------------------------------------------------------
__global__ __launch_bounds__(256) void prep_w_kernel(
    const float* __restrict__ r,       // [E, IN]
    const float* __restrict__ s,       // [E, OUT]
    const float* __restrict__ w,       // [IN, OUT]
    unsigned short* __restrict__ whi,  // [E, OUT, IN] bf16 bits
    unsigned short* __restrict__ wlo)  // [E, OUT, IN] bf16 bits
{
    int t = blockIdx.x * blockDim.x + threadIdx.x;
    if (t >= WSZ) return;
    int o = t % OUTF;
    int i = (t / OUTF) % INF;
    int e = t / (INF * OUTF);
    float wf = r[e * INF + i] * w[i * OUTF + o] * s[e * OUTF + o];
    unsigned short hi = bf16_rne(wf);
    unsigned short lo = bf16_rne(wf - bf16_tof(hi));
    size_t idx = ((size_t)(e * OUTF + o)) * INF + i;   // transposed, k-fast
    whi[idx] = hi;
    wlo[idx] = lo;
}

// ---------------------------------------------------------------------------
// Kernel 2: y = x @ w_e + bias via split-bf16 WMMA.
// Block = 256 threads (8 waves). Wave w handles rows [blk*128 + w*16, +16)
// of expert blockIdx.y, across all 128 output columns (8 accumulator tiles).
// K = 128 split into 4 chunks of 32 (bf16 WMMA depth).
// 3 WMMAs per (kchunk, ntile): hi*hi + lo*hi + hi*lo  (~fp32 accuracy).
// ---------------------------------------------------------------------------
__global__ __launch_bounds__(256) void moe_gemm_kernel(
    const float* __restrict__ x,              // [E*B, IN]
    const float* __restrict__ bias,           // [E, OUT]
    const unsigned short* __restrict__ whi,   // [E, OUT, IN] bf16 bits
    const unsigned short* __restrict__ wlo,   // [E, OUT, IN] bf16 bits
    float* __restrict__ out)                  // [E*B, OUT]
{
    const int e    = blockIdx.y;
    const int wave = threadIdx.x >> 5;
    const int lane = threadIdx.x & 31;
    const int kh   = lane >> 4;        // K-half selector within fragment
    const int ml   = lane & 15;        // row (A) / column (B) within tile

    const long long rowBase =
        (long long)e * BROWS + (long long)blockIdx.x * 128 + wave * 16;

    // A order used here (16x32, same k order as B): lane<16 -> M=lane, K = 0..15 contiguous;
    //                            lane>=16 -> M=lane-16, K = 16..31 contiguous.
    const float* xA = x + (rowBase + ml) * (long long)INF + kh * 16;

    // B order used here (32x16, same k order as A): lane<16 -> N=lane, K=0..15 contiguous;
    //                          lane>=16 -> N=lane-16, K=16..31 contiguous.
    const unsigned short* bhiBase =
        whi + ((size_t)(e * OUTF) + ml) * INF + kh * 16;
    const unsigned short* bloBase =
        wlo + ((size_t)(e * OUTF) + ml) * INF + kh * 16;

    v8f acc[8] = {};

#pragma unroll
    for (int kc = 0; kc < 4; ++kc) {
        const int kb = kc * 32;

        // ---- load & split 16 floats of A for this lane (nontemporal: x is
        // read exactly once) ----
        fx4 q0 = __builtin_nontemporal_load((const fx4*)(xA + kb));
        fx4 q1 = __builtin_nontemporal_load((const fx4*)(xA + kb + 4));
        fx4 q2 = __builtin_nontemporal_load((const fx4*)(xA + kb + 8));
        fx4 q3 = __builtin_nontemporal_load((const fx4*)(xA + kb + 12));
        float a[16];
        a[0]=q0[0];  a[1]=q0[1];  a[2]=q0[2];  a[3]=q0[3];
        a[4]=q1[0];  a[5]=q1[1];  a[6]=q1[2];  a[7]=q1[3];
        a[8]=q2[0];  a[9]=q2[1];  a[10]=q2[2]; a[11]=q2[3];
        a[12]=q3[0]; a[13]=q3[1]; a[14]=q3[2]; a[15]=q3[3];

        v16u uh, ul;
#pragma unroll
        for (int j = 0; j < 16; ++j) {
            unsigned short h = bf16_rne(a[j]);
            uh[j] = h;
            ul[j] = bf16_rne(a[j] - bf16_tof(h));
        }
        v16bf ah = __builtin_bit_cast(v16bf, uh);
        v16bf al = __builtin_bit_cast(v16bf, ul);

        // ---- 8 N-tiles of 16 columns ----
#pragma unroll
        for (int nt = 0; nt < 8; ++nt) {
            const size_t boff = (size_t)nt * 16 * INF + kb;
            v16bf bh = __builtin_bit_cast(v16bf, *(const v16u*)(bhiBase + boff));
            v16bf bl = __builtin_bit_cast(v16bf, *(const v16u*)(bloBase + boff));
            acc[nt] = __builtin_amdgcn_wmma_f32_16x16x32_bf16(
                false, ah, false, bh, (short)0, acc[nt], false, false);
            acc[nt] = __builtin_amdgcn_wmma_f32_16x16x32_bf16(
                false, al, false, bh, (short)0, acc[nt], false, false);
            acc[nt] = __builtin_amdgcn_wmma_f32_16x16x32_bf16(
                false, ah, false, bl, (short)0, acc[nt], false, false);
        }
    }

    // ---- epilogue: C/D layout: VGPR v -> M = v + 8*(lane>=16), N = lane%16.
    // Add bias and stream out (nontemporal: write-once).
    const long long orow = rowBase + kh * 8;
#pragma unroll
    for (int nt = 0; nt < 8; ++nt) {
        const int col = nt * 16 + ml;
        const float bb = bias[e * OUTF + col];
        float* op = out + orow * (long long)OUTF + col;
#pragma unroll
        for (int v = 0; v < 8; ++v) {
            __builtin_nontemporal_store(acc[nt][v] + bb, op + (size_t)v * OUTF);
        }
    }
}

// ---------------------------------------------------------------------------
// Launch: inputs in setup_inputs() order: x, r, s, weight, bias (all fp32).
// d_ws holds the two bf16 split-weight planes (256 KB total).
// ---------------------------------------------------------------------------
extern "C" void kernel_launch(void* const* d_in, const int* in_sizes, int n_in,
                              void* d_out, int out_size, void* d_ws, size_t ws_size,
                              hipStream_t stream) {
    const float* x    = (const float*)d_in[0];
    const float* r    = (const float*)d_in[1];
    const float* s    = (const float*)d_in[2];
    const float* w    = (const float*)d_in[3];
    const float* bias = (const float*)d_in[4];
    float* out = (float*)d_out;

    unsigned short* whi = (unsigned short*)d_ws;
    unsigned short* wlo = whi + (size_t)WSZ;

    prep_w_kernel<<<(WSZ + 255) / 256, 256, 0, stream>>>(r, s, w, whi, wlo);

    dim3 grid(BROWS / 128, NEXP);
    moe_gemm_kernel<<<grid, 256, 0, stream>>>(x, bias, whi, wlo, out);
}
